// AudioSeq2seq_22720376996179
// MI455X (gfx1250) — matrix operand order repaired and checked by reading, not compiled, not run on hardware
//
#include <hip/hip_runtime.h>
#include <hip/hip_bf16.h>

// ---------------------------------------------------------------------------
// Types / WMMA helpers (CDNA5 gfx1250, wave32)
// ---------------------------------------------------------------------------
typedef __attribute__((ext_vector_type(16))) __bf16 v16bf;
typedef __attribute__((ext_vector_type(8)))  float  v8f;

__device__ __forceinline__ __bf16 f2bf(float f) {
  union { float f; unsigned u; } v; v.f = f;
  unsigned r = (v.u + 0x7FFFu + ((v.u >> 16) & 1u)) >> 16;
  union { unsigned short s; __bf16 b; } o; o.s = (unsigned short)r;
  return o.b;
}
__device__ __forceinline__ float bf2f(__bf16 b) {
  union { unsigned short s; __bf16 b; } i; i.b = b;
  union { float f; unsigned u; } v; v.u = ((unsigned)i.s) << 16;
  return v.f;
}
__device__ __forceinline__ float sigmoidf_(float x) { return 1.f / (1.f + __expf(-x)); }

// CDNA5 async memory->LDS (ASYNCcnt tracked, no VGPR round trip).
// VDST = per-lane LDS byte address, VADDR = per-lane 64-bit global address.
__device__ __forceinline__ void async_copy_b128(const void* gsrc, void* lds_dst) {
  unsigned ldsoff = (unsigned)(unsigned long long)lds_dst; // aperture: LDS addr = addr[31:0]
  asm volatile("global_load_async_to_lds_b128 %0, %1, off"
               :: "v"(ldsoff), "v"(gsrc) : "memory");
}
__device__ __forceinline__ void wait_async0() {
  asm volatile("s_wait_asynccnt 0x0" ::: "memory");
}

// A-fragment (16x32 bf16), rows relative to `base`, row stride ldk elements.
// Lane l<16:  M=l,    K = 0..15
// Lane l>=16: M=l-16, K = 16..31
__device__ __forceinline__ v16bf afrag_ld(const __bf16* base, int ldk) {
  int lane = threadIdx.x & 31;
  int m = lane & 15;
  int hi = lane >> 4;
  const __bf16* p = base + (long)m * ldk + hi * 16;
  v16bf r;
  *(float4*)&r       = *(const float4*)p;
  *((float4*)&r + 1) = *(const float4*)(p + 8);
  return r;
}

// B-fragment (32x16 bf16) from pre-packed weights: per (ktile,ntile) tile,
// lane l holds 16 contiguous bf16 = B[kt*32 + 16*(l>=16) + j][nt*16 + (l&15)].
__device__ __forceinline__ v16bf bfrag_ld(const __bf16* Bp, long tile) {
  int lane = threadIdx.x & 31;
  const __bf16* p = Bp + ((tile << 5) + lane) * 16;
  v16bf r;
  *(float4*)&r       = *(const float4*)p;
  *((float4*)&r + 1) = *(const float4*)(p + 8);
  return r;
}

__device__ __forceinline__ v8f wmma_bf16(v16bf a, v16bf b, v8f c) {
  return __builtin_amdgcn_wmma_f32_16x16x32_bf16(false, a, false, b, (short)0, c,
                                                 false, false);
}

// ---------------------------------------------------------------------------
// Model dims
// ---------------------------------------------------------------------------
#define BB   32
#define T1   1600
#define T2   800
#define HD_  256
#define HH   512
#define EMB_ 512
#define TD_  161     // decoder steps (T+1)
#define AA   128
#define NFF  32
#define KK   31
#define NSYM_ 101

// ---------------------------------------------------------------------------
// Weight packing: W is [N][K] row-major f32 (out,in). We compute x @ W^T, so
// packed B[k][n] = W[n][k], laid out fragment-ready (see bfrag_ld).
// ---------------------------------------------------------------------------
__global__ void pack_b_kernel(const float* __restrict__ W, __bf16* __restrict__ P,
                              int K, int N, int Kp, int Np) {
  int ntiles = Np >> 4;
  long total = (long)Kp * Np;
  for (long idx = (long)blockIdx.x * blockDim.x + threadIdx.x; idx < total;
       idx += (long)gridDim.x * blockDim.x) {
    long j    = idx & 15;
    long lane = (idx >> 4) & 31;
    long tile = idx >> 9;
    long kt = tile / ntiles, nt = tile % ntiles;
    long k = kt * 32 + ((lane >= 16) ? 16 : 0) + j;
    long n = nt * 16 + (lane & 15);
    float v = (k < K && n < N) ? W[n * (long)K + k] : 0.f;
    P[idx] = f2bf(v);
  }
}

__global__ void k_memlen(const int* __restrict__ len, int* __restrict__ mlen) {
  int b = threadIdx.x;
  if (b < BB) mlen[b] = (len[b] + 1) >> 1;
}

// mel [B][80][T1] f32 -> x1f/x1r [T1][B][96] bf16 (K padded 80->96, bwd reversed)
__global__ void build_x1(const float* __restrict__ mel, const int* __restrict__ len,
                         __bf16* __restrict__ x1f, __bf16* __restrict__ x1r) {
  const long total = 2L * T1 * BB * 96;
  for (long i = (long)blockIdx.x * blockDim.x + threadIdx.x; i < total;
       i += (long)gridDim.x * blockDim.x) {
    long r = i;
    int k = (int)(r % 96); r /= 96;
    int b = (int)(r % BB); r /= BB;
    int t = (int)(r % T1); r /= T1;
    int dir = (int)r;
    int ts = t;
    if (dir) { int L = len[b]; ts = L - 1 - t; if (ts < 0) ts = 0; }
    float v = (k < 80) ? mel[((long)b * 80 + k) * T1 + ts] : 0.f;
    (dir ? x1r : x1f)[((long)t * BB + b) * 96 + k] = f2bf(v);
  }
}

// dinp [161][B][512] bf16: t=0 start_embedding, t>0 decoder_inputs[:,:,t-1]
__global__ void build_dinp(const float* __restrict__ start,
                           const float* __restrict__ decin,
                           __bf16* __restrict__ dinp) {
  const long total = (long)TD_ * BB * EMB_;
  for (long i = (long)blockIdx.x * blockDim.x + threadIdx.x; i < total;
       i += (long)gridDim.x * blockDim.x) {
    int e = (int)(i % EMB_);
    int b = (int)((i / EMB_) % BB);
    int t = (int)(i / ((long)EMB_ * BB));
    float v = (t == 0) ? start[b * EMB_ + e]
                       : decin[((long)b * EMB_ + e) * (TD_ - 1) + (t - 1)];
    dinp[i] = f2bf(v);
  }
}

// out1 reshape + reversal for layer2: x2f/x2r [T2][B][1024] bf16
__global__ void build_x2(const __bf16* __restrict__ y1f, const __bf16* __restrict__ y1r,
                         const int* __restrict__ len, const int* __restrict__ mlen,
                         __bf16* __restrict__ x2f, __bf16* __restrict__ x2r) {
  const long total = 2L * T2 * BB * 1024;
  for (long i = (long)blockIdx.x * blockDim.x + threadIdx.x; i < total;
       i += (long)gridDim.x * blockDim.x) {
    long r = i;
    int c  = (int)(r % 1024); r /= 1024;
    int b  = (int)(r % BB);   r /= BB;
    int t2 = (int)(r % T2);   r /= T2;
    int dir = (int)r;
    int trow = t2;
    if (dir) { int M = mlen[b]; trow = M - 1 - t2; if (trow < 0) trow = 0; }
    int tau = 2 * trow + (c >= 512);
    int cc = c & 511;
    float v;
    if (cc < HD_) {
      v = bf2f(y1f[((long)tau * BB + b) * HD_ + cc]);
    } else {
      int L = len[b];
      if (tau < L) v = bf2f(y1r[((long)(L - 1 - tau) * BB + b) * HD_ + (cc - HD_)]);
      else v = 0.f;
    }
    (dir ? x2r : x2f)[((long)t2 * BB + b) * 1024 + c] = f2bf(v);
  }
}

// memory [B][T2][512] bf16 from layer-2 outputs
__global__ void build_mem(const __bf16* __restrict__ y2f, const __bf16* __restrict__ y2r,
                          const int* __restrict__ mlen, __bf16* __restrict__ memb) {
  const long total = (long)BB * T2 * HH;
  for (long i = (long)blockIdx.x * blockDim.x + threadIdx.x; i < total;
       i += (long)gridDim.x * blockDim.x) {
    int c = (int)(i % HH);
    int t = (int)((i / HH) % T2);
    int b = (int)(i / ((long)HH * T2));
    float v;
    if (c < HD_) v = bf2f(y2f[((long)t * BB + b) * HD_ + c]);
    else {
      int M = mlen[b];
      if (t < M) v = bf2f(y2r[((long)(M - 1 - t) * BB + b) * HD_ + (c - HD_)]);
      else v = 0.f;
    }
    memb[i] = f2bf(v);
  }
}

// ---------------------------------------------------------------------------
// Parallel WMMA GEMM: pm[M][128] = memory[M][512] @ mW^T (packed), M = 25600
// one wave per 16-row M tile, 8 N tiles resident as accumulators
// ---------------------------------------------------------------------------
__global__ void pm_gemm(const __bf16* __restrict__ Amat, const __bf16* __restrict__ Bp,
                        float* __restrict__ C, int Mtiles) {
  int wv = threadIdx.x >> 5, lane = threadIdx.x & 31;
  int mt = blockIdx.x * (blockDim.x >> 5) + wv;
  if (mt >= Mtiles) return;
  v8f acc[8];
#pragma unroll
  for (int n = 0; n < 8; ++n) { v8f z = {}; acc[n] = z; }
  const __bf16* Abase = Amat + (long)mt * 16 * HH;
  for (int kc = 0; kc < 16; ++kc) {
    v16bf a = afrag_ld(Abase + kc * 32, HH);
#pragma unroll
    for (int nt = 0; nt < 8; ++nt) {
      v16bf b = bfrag_ld(Bp, (long)kc * 8 + nt);
      acc[nt] = wmma_bf16(a, b, acc[nt]);
    }
  }
#pragma unroll
  for (int nt = 0; nt < 8; ++nt)
#pragma unroll
    for (int r = 0; r < 8; ++r) {
      int mrow = mt * 16 + r + ((lane >= 16) ? 8 : 0);
      C[(long)mrow * AA + nt * 16 + (lane & 15)] = acc[nt][r];
    }
}

// ---------------------------------------------------------------------------
// Persistent masked LSTM (one block per direction, 32 waves).
// gates[32,1024] = x_t[32,Kx] @ WihT + h[32,256] @ WhhT + b
// wave w -> rowtile = w>>4, gatecol = w&15 (owns i/f/g/o of one 16x16 patch).
// x_t is double-buffered in LDS; the t+1 tile is fetched with
// GLOBAL_LOAD_ASYNC_TO_LDS_B128 and overlapped with the step-t WMMA chain.
// ---------------------------------------------------------------------------
__global__ void lstm_kernel(const __bf16* __restrict__ xf, const __bf16* __restrict__ xr,
                            const __bf16* __restrict__ wihf, const __bf16* __restrict__ wihr,
                            const __bf16* __restrict__ whhf, const __bf16* __restrict__ whhr,
                            const float* __restrict__ bfw, const float* __restrict__ bbw,
                            const int* __restrict__ lengths,
                            __bf16* __restrict__ yf, __bf16* __restrict__ yr,
                            int T, int Kx) {
  const int dir = blockIdx.x;
  const __bf16* x   = dir ? xr   : xf;
  const __bf16* wih = dir ? wihr : wihf;
  const __bf16* whh = dir ? whhr : whhf;
  const float*  bia = dir ? bbw  : bfw;
  __bf16* y = dir ? yr : yf;

  extern __shared__ char smem[];
  __bf16* xs0 = (__bf16*)smem;           // [32][Kx] double-buffered
  __bf16* xs1 = xs0 + 32 * Kx;
  __bf16* hs  = xs1 + 32 * Kx;           // [32][256]

  const int tid = threadIdx.x;
  const int lane = tid & 31;
  const int wv = tid >> 5;
  const int rowtile = wv >> 4;
  const int gatecol = wv & 15;
  const int ncol = gatecol * 16 + (lane & 15);

  for (int i = tid; i < 32 * HD_; i += blockDim.x) hs[i] = f2bf(0.f);

  int blen[8];
  float bgate[4];
#pragma unroll
  for (int r = 0; r < 8; ++r)
    blen[r] = lengths[rowtile * 16 + r + ((lane >= 16) ? 8 : 0)];
#pragma unroll
  for (int g = 0; g < 4; ++g) bgate[g] = bia[g * HD_ + ncol];

  v8f cst = {};
  const int  KCX = Kx >> 5;
  const long xstep = 32L * Kx;
  const int  bytes = 32 * Kx * 2;

  // prologue: async-stage x_0 into xs0
  {
    const char* src = (const char*)x;
    char* dst = (char*)xs0;
    for (int o = tid * 16; o < bytes; o += blockDim.x * 16)
      async_copy_b128(src + o, dst + o);
  }

  for (int t = 0; t < T; ++t) {
    wait_async0();        // x_t resident in LDS
    __syncthreads();      // visible to all waves; prev-step readers done

    __bf16* xs = (t & 1) ? xs1 : xs0;
    if (t + 1 < T) {      // overlap next tile's DMA with this step's WMMAs
      const char* src = (const char*)(x + (long)(t + 1) * xstep);
      char* dst = (char*)((t & 1) ? xs0 : xs1);
      for (int o = tid * 16; o < bytes; o += blockDim.x * 16)
        async_copy_b128(src + o, dst + o);
    }

    v8f acc[4];
#pragma unroll
    for (int g = 0; g < 4; ++g) {
      v8f z = {};
#pragma unroll
      for (int r = 0; r < 8; ++r) z[r] = bgate[g];
      acc[g] = z;
    }
    for (int kc = 0; kc < KCX; ++kc) {
      v16bf a = afrag_ld(xs + rowtile * 16 * Kx + kc * 32, Kx);
#pragma unroll
      for (int g = 0; g < 4; ++g) {
        v16bf b = bfrag_ld(wih, (long)kc * 64 + g * 16 + gatecol);
        acc[g] = wmma_bf16(a, b, acc[g]);
      }
    }
    for (int kc = 0; kc < 8; ++kc) {
      v16bf a = afrag_ld(hs + rowtile * 16 * HD_ + kc * 32, HD_);
#pragma unroll
      for (int g = 0; g < 4; ++g) {
        v16bf b = bfrag_ld(whh, (long)kc * 64 + g * 16 + gatecol);
        acc[g] = wmma_bf16(a, b, acc[g]);
      }
    }
    __syncthreads();   // all h reads done before h is rewritten

#pragma unroll
    for (int r = 0; r < 8; ++r) {
      int b = rowtile * 16 + r + ((lane >= 16) ? 8 : 0);
      long yi = (long)t * 32 * HD_ + (long)b * HD_ + ncol;
      if (t < blen[r]) {
        float iv = sigmoidf_(acc[0][r]);
        float fv = sigmoidf_(acc[1][r]);
        float gv = tanhf(acc[2][r]);
        float ov = sigmoidf_(acc[3][r]);
        float cn = fv * cst[r] + iv * gv;
        float hn = ov * tanhf(cn);
        cst[r] = cn;
        hs[b * HD_ + ncol] = f2bf(hn);
        y[yi] = f2bf(hn);
      } else {
        y[yi] = f2bf(0.f);
      }
    }
  }
}

// ---------------------------------------------------------------------------
// Persistent decoder: LSTM cell (WMMA) + location attention + projections.
// gates: [32,1536] x [1536,2048]; wave w -> rowtile=w>>4, gatecols 2(w&15),+1
// ---------------------------------------------------------------------------
__global__ void decoder_kernel(const __bf16* __restrict__ dinp,
                               const __bf16* __restrict__ wihp,
                               const __bf16* __restrict__ whhp,
                               const float* __restrict__ dec_b,
                               const __bf16* __restrict__ php,
                               const float* __restrict__ ph_b,
                               const float* __restrict__ qW,
                               const float* __restrict__ locconvW,
                               const float* __restrict__ locW,
                               const float* __restrict__ vW,
                               const float* __restrict__ psW,
                               const float* __restrict__ ps_b,
                               const __bf16* __restrict__ memb,
                               const float* __restrict__ pm,
                               const int* __restrict__ mlen,
                               float* __restrict__ awbuf, float* __restrict__ awcbuf,
                               float* __restrict__ ebuf,
                               float* __restrict__ out_hid,
                               float* __restrict__ out_logit,
                               float* __restrict__ out_align, int T) {
  extern __shared__ char smem[];
  __bf16* cellin = (__bf16*)smem;          // [32][1024]  [dinp|ctx] then [h|ctx]
  __bf16* hbuf   = cellin + 32 * 1024;     // [32][512]
  __bf16* hidb   = hbuf + 32 * HH;         // [32][512]
  float*  qs     = (float*)(hidb + 32 * HH); // [32][128]

  const int tid = threadIdx.x, lane = tid & 31, wv = tid >> 5;
  const int rowtile = wv >> 4;
  const int gcA = (wv & 15) * 2, gcB = gcA + 1;

  for (int i = tid; i < 32 * 1024; i += blockDim.x) cellin[i] = f2bf(0.f);
  for (int i = tid; i < 32 * HH; i += blockDim.x) hbuf[i] = f2bf(0.f);
  for (int i = tid; i < BB * T2; i += blockDim.x) { awbuf[i] = 0.f; awcbuf[i] = 0.f; }

  v8f cA = {}, cB = {};
  float bgA[4], bgB[4];
#pragma unroll
  for (int g = 0; g < 4; ++g) {
    bgA[g] = dec_b[g * HH + gcA * 16 + (lane & 15)];
    bgB[g] = dec_b[g * HH + gcB * 16 + (lane & 15)];
  }
  int mrow[8];
#pragma unroll
  for (int r = 0; r < 8; ++r) mrow[r] = rowtile * 16 + r + ((lane >= 16) ? 8 : 0);

  __syncthreads();

  for (int t = 0; t < T; ++t) {
    // ---- async-stage decoder input into cellin[:, :512] (ctx half kept) ----
    {
      const char* src = (const char*)(dinp + (long)t * 32 * EMB_);
      for (int i = tid * 8; i < 32 * EMB_; i += blockDim.x * 8) {
        int b = i >> 9, e = i & 511;
        async_copy_b128(src + (long)i * 2, (char*)(cellin + b * 1024 + e));
      }
      wait_async0();
    }
    __syncthreads();

    // ---- gates GEMM: K = 1024 (cellin) + 512 (h) ----
    v8f aA[4], aB[4];
#pragma unroll
    for (int g = 0; g < 4; ++g) {
      v8f z0 = {}, z1 = {};
#pragma unroll
      for (int r = 0; r < 8; ++r) { z0[r] = bgA[g]; z1[r] = bgB[g]; }
      aA[g] = z0; aB[g] = z1;
    }
    for (int kc = 0; kc < 32; ++kc) {
      v16bf a = afrag_ld(cellin + rowtile * 16 * 1024 + kc * 32, 1024);
#pragma unroll
      for (int g = 0; g < 4; ++g) {
        v16bf b0 = bfrag_ld(wihp, (long)kc * 128 + g * 32 + gcA);
        aA[g] = wmma_bf16(a, b0, aA[g]);
        v16bf b1 = bfrag_ld(wihp, (long)kc * 128 + g * 32 + gcB);
        aB[g] = wmma_bf16(a, b1, aB[g]);
      }
    }
    for (int kc = 0; kc < 16; ++kc) {
      v16bf a = afrag_ld(hbuf + rowtile * 16 * HH + kc * 32, HH);
#pragma unroll
      for (int g = 0; g < 4; ++g) {
        v16bf b0 = bfrag_ld(whhp, (long)kc * 128 + g * 32 + gcA);
        aA[g] = wmma_bf16(a, b0, aA[g]);
        v16bf b1 = bfrag_ld(whhp, (long)kc * 128 + g * 32 + gcB);
        aB[g] = wmma_bf16(a, b1, aB[g]);
      }
    }
    __syncthreads();

    // ---- LSTM pointwise (wave-local i/f/g/o), h -> LDS ----
#pragma unroll
    for (int r = 0; r < 8; ++r) {
      {
        float iv = sigmoidf_(aA[0][r]), fv = sigmoidf_(aA[1][r]);
        float gv = tanhf(aA[2][r]), ov = sigmoidf_(aA[3][r]);
        float cn = fv * cA[r] + iv * gv; cA[r] = cn;
        hbuf[mrow[r] * HH + gcA * 16 + (lane & 15)] = f2bf(ov * tanhf(cn));
      }
      {
        float iv = sigmoidf_(aB[0][r]), fv = sigmoidf_(aB[1][r]);
        float gv = tanhf(aB[2][r]), ov = sigmoidf_(aB[3][r]);
        float cn = fv * cB[r] + iv * gv; cB[r] = cn;
        hbuf[mrow[r] * HH + gcB * 16 + (lane & 15)] = f2bf(ov * tanhf(cn));
      }
    }
    __syncthreads();

    // ---- query projection q = h @ qW^T ----
    for (int i = tid; i < BB * AA; i += blockDim.x) {
      int b = i >> 7, a = i & 127;
      const float* w = qW + (long)a * HH;
      const __bf16* hh = hbuf + b * HH;
      float s = 0.f;
      for (int k = 0; k < HH; ++k) s += bf2f(hh[k]) * w[k];
      qs[i] = s;
    }
    __syncthreads();

    // ---- attention energies e[b,t'] ----
    for (int p = tid; p < BB * T2; p += blockDim.x) {
      int b = p / T2, tp = p % T2;
      float ev;
      if (tp >= mlen[b]) {
        ev = -1e9f;
      } else {
        float lv[NFF];
#pragma unroll
        for (int f = 0; f < NFF; ++f) lv[f] = 0.f;
        const float* awr  = awbuf + b * T2;
        const float* awcr = awcbuf + b * T2;
        for (int k = 0; k < KK; ++k) {
          int ts = tp + k - 15;
          if (ts >= 0 && ts < T2) {
            float a0 = awr[ts], a1 = awcr[ts];
#pragma unroll
            for (int f = 0; f < NFF; ++f)
              lv[f] += locconvW[f * 62 + k] * a0 + locconvW[f * 62 + KK + k] * a1;
          }
        }
        const float* pmrow = pm + ((long)b * T2 + tp) * AA;
        float s = 0.f;
        for (int a = 0; a < AA; ++a) {
          float l = 0.f;
#pragma unroll
          for (int f = 0; f < NFF; ++f) l += locW[a * NFF + f] * lv[f];
          s += vW[a] * tanhf(qs[b * AA + a] + l + pmrow[a]);
        }
        ev = s;
      }
      ebuf[p] = ev;
    }
    __syncthreads();

    // ---- softmax per batch (one wave per batch, wave32 shuffles) ----
    {
      int b = wv;
      const float* er = ebuf + b * T2;
      float mx = -3.4e38f;
      for (int i = lane; i < T2; i += 32) mx = fmaxf(mx, er[i]);
#pragma unroll
      for (int o = 16; o >= 1; o >>= 1) mx = fmaxf(mx, __shfl_xor(mx, o, 32));
      float sm = 0.f;
      for (int i = lane; i < T2; i += 32) sm += __expf(er[i] - mx);
#pragma unroll
      for (int o = 16; o >= 1; o >>= 1) sm += __shfl_xor(sm, o, 32);
      float inv = 1.f / sm;
      for (int i = lane; i < T2; i += 32) {
        float a = __expf(er[i] - mx) * inv;
        awbuf[b * T2 + i] = a;
        awcbuf[b * T2 + i] += a;
        out_align[((long)b * T + t) * T2 + i] = a;
      }
    }
    __syncthreads();

    // ---- context ctx[b,d] = sum_t aw * memory; store bf16 in cellin ctx half
    for (int i = tid; i < BB * HH; i += blockDim.x) {
      int b = i >> 9, d = i & 511;
      const float* ar = awbuf + b * T2;
      const __bf16* mr = memb + (long)b * T2 * HH + d;
      int lim = mlen[b];
      float s = 0.f;
      for (int tt = 0; tt < lim; ++tt) s += ar[tt] * bf2f(mr[(long)tt * HH]);
      cellin[b * 1024 + HH + d] = f2bf(s);
    }
    // copy h into cellin[:, :512]: hid GEMM input = [h | ctx]
    for (int i = tid; i < BB * HH; i += blockDim.x) {
      int b = i >> 9, d = i & 511;
      cellin[b * 1024 + d] = hbuf[i];
    }
    __syncthreads();

    // ---- hid = relu([h|ctx] @ phW^T + ph_b)  (WMMA, 2 tiles per wave) ----
    {
      int rt = wv >> 4;
      int ntA2 = (wv & 15) * 2, ntB2 = ntA2 + 1;
      v8f hA = {}, hB = {};
      float bA2 = ph_b[ntA2 * 16 + (lane & 15)];
      float bB2 = ph_b[ntB2 * 16 + (lane & 15)];
#pragma unroll
      for (int r = 0; r < 8; ++r) { hA[r] = bA2; hB[r] = bB2; }
      for (int kc = 0; kc < 32; ++kc) {
        v16bf a = afrag_ld(cellin + rt * 16 * 1024 + kc * 32, 1024);
        v16bf b0 = bfrag_ld(php, (long)kc * 32 + ntA2);
        hA = wmma_bf16(a, b0, hA);
        v16bf b1 = bfrag_ld(php, (long)kc * 32 + ntB2);
        hB = wmma_bf16(a, b1, hB);
      }
#pragma unroll
      for (int r = 0; r < 8; ++r) {
        int b = rt * 16 + r + ((lane >= 16) ? 8 : 0);
        int n0 = ntA2 * 16 + (lane & 15), n1 = ntB2 * 16 + (lane & 15);
        float v0 = fmaxf(hA[r], 0.f), v1 = fmaxf(hB[r], 0.f);
        out_hid[((long)b * T + t) * HH + n0] = v0;
        out_hid[((long)b * T + t) * HH + n1] = v1;
        hidb[b * HH + n0] = f2bf(v0);
        hidb[b * HH + n1] = f2bf(v1);
      }
    }
    __syncthreads();

    // ---- logits = hid @ psW^T + ps_b ----
    for (int i = tid; i < BB * NSYM_; i += blockDim.x) {
      int b = i / NSYM_, o = i % NSYM_;
      const float* w = psW + (long)o * HH;
      const __bf16* hh = hidb + b * HH;
      float s = ps_b[o];
      for (int k = 0; k < HH; ++k) s += bf2f(hh[k]) * w[k];
      out_logit[((long)b * T + t) * NSYM_ + o] = s;
    }
    __syncthreads();
  }
}

// ---------------------------------------------------------------------------
// Host launcher
// ---------------------------------------------------------------------------
extern "C" void kernel_launch(void* const* d_in, const int* in_sizes, int n_in,
                              void* d_out, int out_size, void* d_ws, size_t ws_size,
                              hipStream_t stream) {
  (void)in_sizes; (void)n_in; (void)out_size; (void)ws_size;
  const float* mel   = (const float*)d_in[0];
  const int*   lens  = (const int*)d_in[1];
  const float* decin = (const float*)d_in[2];
  const float* start = (const float*)d_in[3];
  const float *l1f_Wih = (const float*)d_in[4],  *l1f_Whh = (const float*)d_in[5],  *l1f_b = (const float*)d_in[6];
  const float *l1b_Wih = (const float*)d_in[7],  *l1b_Whh = (const float*)d_in[8],  *l1b_b = (const float*)d_in[9];
  const float *l2f_Wih = (const float*)d_in[10], *l2f_Whh = (const float*)d_in[11], *l2f_b = (const float*)d_in[12];
  const float *l2b_Wih = (const float*)d_in[13], *l2b_Whh = (const float*)d_in[14], *l2b_b = (const float*)d_in[15];
  const float *dec_Wih = (const float*)d_in[16], *dec_Whh = (const float*)d_in[17], *dec_b = (const float*)d_in[18];
  const float *q_W = (const float*)d_in[19], *m_W = (const float*)d_in[20];
  const float *loc_conv_W = (const float*)d_in[21], *loc_W = (const float*)d_in[22], *v_W = (const float*)d_in[23];
  const float *ph_W = (const float*)d_in[24], *ph_b = (const float*)d_in[25];
  const float *ps_W = (const float*)d_in[26], *ps_b = (const float*)d_in[27];

  char* ws = (char*)d_ws;
  size_t off = 0;
  auto alloc = [&](size_t bytes) -> char* {
    char* p = ws + off;
    off = (off + bytes + 255) & ~(size_t)255;
    return p;
  };

  // packed weights (bf16, fragment-ready)
  __bf16* p_l1f_wih = (__bf16*)alloc((size_t)96 * 1024 * 2);
  __bf16* p_l1b_wih = (__bf16*)alloc((size_t)96 * 1024 * 2);
  __bf16* p_l1f_whh = (__bf16*)alloc((size_t)256 * 1024 * 2);
  __bf16* p_l1b_whh = (__bf16*)alloc((size_t)256 * 1024 * 2);
  __bf16* p_l2f_wih = (__bf16*)alloc((size_t)1024 * 1024 * 2);
  __bf16* p_l2b_wih = (__bf16*)alloc((size_t)1024 * 1024 * 2);
  __bf16* p_l2f_whh = (__bf16*)alloc((size_t)256 * 1024 * 2);
  __bf16* p_l2b_whh = (__bf16*)alloc((size_t)256 * 1024 * 2);
  __bf16* p_dec_wih = (__bf16*)alloc((size_t)1024 * 2048 * 2);
  __bf16* p_dec_whh = (__bf16*)alloc((size_t)512 * 2048 * 2);
  __bf16* p_ph      = (__bf16*)alloc((size_t)1024 * 512 * 2);
  __bf16* p_mw      = (__bf16*)alloc((size_t)512 * 128 * 2);

  // activations
  __bf16* x1f = (__bf16*)alloc((size_t)T1 * BB * 96 * 2);
  __bf16* x1r = (__bf16*)alloc((size_t)T1 * BB * 96 * 2);
  __bf16* y1f = (__bf16*)alloc((size_t)T1 * BB * HD_ * 2);
  __bf16* y1r = (__bf16*)alloc((size_t)T1 * BB * HD_ * 2);
  __bf16* x2f = (__bf16*)alloc((size_t)T2 * BB * 1024 * 2);
  __bf16* x2r = (__bf16*)alloc((size_t)T2 * BB * 1024 * 2);
  __bf16* y2f = (__bf16*)alloc((size_t)T2 * BB * HD_ * 2);
  __bf16* y2r = (__bf16*)alloc((size_t)T2 * BB * HD_ * 2);
  __bf16* memb = (__bf16*)alloc((size_t)BB * T2 * HH * 2);
  float*  pm   = (float*)alloc((size_t)BB * T2 * AA * 4);
  __bf16* dinp = (__bf16*)alloc((size_t)TD_ * BB * EMB_ * 2);
  int*    mlen = (int*)alloc(BB * sizeof(int));
  float*  awbuf  = (float*)alloc((size_t)BB * T2 * 4);
  float*  awcbuf = (float*)alloc((size_t)BB * T2 * 4);
  float*  ebuf   = (float*)alloc((size_t)BB * T2 * 4);

  // ---- weight packing ----
  pack_b_kernel<<<512, 256, 0, stream>>>(l1f_Wih, p_l1f_wih, 80, 1024, 96, 1024);
  pack_b_kernel<<<512, 256, 0, stream>>>(l1b_Wih, p_l1b_wih, 80, 1024, 96, 1024);
  pack_b_kernel<<<512, 256, 0, stream>>>(l1f_Whh, p_l1f_whh, 256, 1024, 256, 1024);
  pack_b_kernel<<<512, 256, 0, stream>>>(l1b_Whh, p_l1b_whh, 256, 1024, 256, 1024);
  pack_b_kernel<<<512, 256, 0, stream>>>(l2f_Wih, p_l2f_wih, 1024, 1024, 1024, 1024);
  pack_b_kernel<<<512, 256, 0, stream>>>(l2b_Wih, p_l2b_wih, 1024, 1024, 1024, 1024);
  pack_b_kernel<<<512, 256, 0, stream>>>(l2f_Whh, p_l2f_whh, 256, 1024, 256, 1024);
  pack_b_kernel<<<512, 256, 0, stream>>>(l2b_Whh, p_l2b_whh, 256, 1024, 256, 1024);
  pack_b_kernel<<<512, 256, 0, stream>>>(dec_Wih, p_dec_wih, 1024, 2048, 1024, 2048);
  pack_b_kernel<<<512, 256, 0, stream>>>(dec_Whh, p_dec_whh, 512, 2048, 512, 2048);
  pack_b_kernel<<<512, 256, 0, stream>>>(ph_W,    p_ph,      1024, 512, 1024, 512);
  pack_b_kernel<<<512, 256, 0, stream>>>(m_W,     p_mw,      512, 128, 512, 128);

  // ---- input prep ----
  k_memlen<<<1, 32, 0, stream>>>(lens, mlen);
  build_x1<<<2048, 256, 0, stream>>>(mel, lens, x1f, x1r);
  build_dinp<<<1024, 256, 0, stream>>>(start, decin, dinp);

  // ---- encoder layer 1 (fwd+bwd concurrently, double-buffered async x) ----
  size_t lds1 = (size_t)(2 * 32 * 96 + 32 * HD_) * 2;
  lstm_kernel<<<2, 1024, lds1, stream>>>(x1f, x1r, p_l1f_wih, p_l1b_wih,
                                         p_l1f_whh, p_l1b_whh, l1f_b, l1b_b,
                                         lens, y1f, y1r, T1, 96);

  build_x2<<<4096, 256, 0, stream>>>(y1f, y1r, lens, mlen, x2f, x2r);

  // ---- encoder layer 2 ----
  size_t lds2 = (size_t)(2 * 32 * 1024 + 32 * HD_) * 2;
  lstm_kernel<<<2, 1024, lds2, stream>>>(x2f, x2r, p_l2f_wih, p_l2b_wih,
                                         p_l2f_whh, p_l2b_whh, l2f_b, l2b_b,
                                         mlen, y2f, y2r, T2, 1024);

  build_mem<<<2048, 256, 0, stream>>>(y2f, y2r, mlen, memb);

  // ---- processed_memory = memory @ mW^T  (parallel WMMA, 1600 M tiles) ----
  pm_gemm<<<50, 1024, 0, stream>>>(memb, p_mw, pm, (BB * T2) / 16);

  // ---- decoder ----
  float* out_hid   = (float*)d_out;
  float* out_logit = out_hid + (long)BB * TD_ * HH;
  float* out_align = out_logit + (long)BB * TD_ * NSYM_;
  size_t ldsd = (size_t)(32 * 1024 + 32 * HH + 32 * HH) * 2 + (size_t)32 * AA * 4;
  decoder_kernel<<<1, 1024, ldsd, stream>>>(dinp, p_dec_wih, p_dec_whh, dec_b,
                                            p_ph, ph_b, q_W, loc_conv_W, loc_W, v_W,
                                            ps_W, ps_b, memb, pm, mlen,
                                            awbuf, awcbuf, ebuf,
                                            out_hid, out_logit, out_align, TD_);
}
